// BiDAFNet_13262859010787
// MI455X (gfx1250) — matrix operand order repaired and checked by reading, not compiled, not run on hardware
//
#include <hip/hip_runtime.h>

// ---------------- problem constants ----------------
#define B_N   8
#define Q_L   512
#define C_L   2048
#define H_D   768
#define OUT_D 300
#define WR_ROWS 320          // OUT padded to 320 (5 waves * 64 cols)
#define K4  (4 * H_D)        // 3072

typedef __bf16 v16bf __attribute__((ext_vector_type(16)));
typedef float  v8f   __attribute__((ext_vector_type(8)));
typedef unsigned short u16x16 __attribute__((ext_vector_type(16)));

// ---------------- bf16 helpers ----------------
__device__ __forceinline__ unsigned short f2bf(float f) {
    unsigned u = __builtin_bit_cast(unsigned, f);
    unsigned r = u + 0x7FFFu + ((u >> 16) & 1u);   // round-to-nearest-even
    return (unsigned short)(r >> 16);
}
__device__ __forceinline__ float bf2f(unsigned short h) {
    unsigned u = ((unsigned)h) << 16;
    return __builtin_bit_cast(float, u);
}

// ---------------- WMMA fragment loaders ----------------
// A fragment 16x32 (MxK): lane m = lane&15; K chunk [sel*16 .. +15]
__device__ __forceinline__ v16bf load_frag_a(const unsigned short* p, int ld,
                                             int kBase, int lane) {
    int row = lane & 15, sel = lane >> 4;
    const unsigned short* r0 = p + (size_t)row * ld + kBase + sel * 16;
    u16x16 u;
#pragma unroll
    for (int i = 0; i < 16; ++i) u[i] = r0[i];
    return __builtin_bit_cast(v16bf, u);
}

// B fragment 32x16 (KxN): p points at B^T row-major (row n = column n of B, contiguous K)
__device__ __forceinline__ v16bf load_frag_b(const unsigned short* p, int ld,
                                             int kBase, int lane) {
    int col = lane & 15, sel = lane >> 4;
    const unsigned short* r0 = p + (size_t)col * ld + kBase + sel * 16;
    u16x16 u;
#pragma unroll
    for (int i = 0; i < 16; ++i) u[i] = r0[i];
    return __builtin_bit_cast(v16bf, u);
}

__device__ __forceinline__ v8f wmma_bf16(v16bf a, v16bf b, v8f c) {
    return __builtin_amdgcn_wmma_f32_16x16x32_bf16(false, a, false, b,
                                                   (short)0, c, false, false);
}

// ---------------- prep kernels ----------------
// emb2 -> e2b (bf16) and e2w = bf16(emb2 * w_cq) in one pass
__global__ void k_prep_e2(const float* __restrict__ emb2,
                          const float* __restrict__ w_cq,
                          unsigned short* __restrict__ e2b,
                          unsigned short* __restrict__ e2w, long long n) {
    long long i = (long long)blockIdx.x * blockDim.x + threadIdx.x;
    if (i >= n) return;
    float v = emb2[i];
    int h = (int)(i % H_D);
    e2b[i] = f2bf(v);
    e2w[i] = f2bf(v * w_cq[h]);
}

__global__ void k_cvt_bf16(const float* __restrict__ src,
                           unsigned short* __restrict__ dst, long long n) {
    long long i = (long long)blockIdx.x * blockDim.x + threadIdx.x;
    if (i < n) dst[i] = f2bf(src[i]);
}

// emb1 (B,Q,H) f32 -> e1t (B,H,Q) bf16  (coalesced writes)
__global__ void k_e1_transpose(const float* __restrict__ emb1,
                               unsigned short* __restrict__ e1t) {
    long long i = (long long)blockIdx.x * blockDim.x + threadIdx.x;
    long long n = (long long)B_N * H_D * Q_L;
    if (i >= n) return;
    int q = (int)(i % Q_L);
    int h = (int)((i / Q_L) % H_D);
    int b = (int)(i / ((long long)Q_L * H_D));
    e1t[i] = f2bf(emb1[((size_t)b * Q_L + q) * H_D + h]);
}

// w_red (OUT,4H) f32 -> wrb (WR_ROWS,4H) bf16, zero padded rows
__global__ void k_cvt_wred(const float* __restrict__ w_red,
                           unsigned short* __restrict__ wrb) {
    long long i = (long long)blockIdx.x * blockDim.x + threadIdx.x;
    long long n = (long long)WR_ROWS * K4;
    if (i >= n) return;
    int o = (int)(i / K4);
    unsigned short v = 0;
    if (o < OUT_D) v = f2bf(w_red[i]);
    wrb[i] = v;
}

// one wave per row: out[row] = dot(X[row,:], w) + bias[0]
__global__ void k_rowdot(const float* __restrict__ X, const float* __restrict__ w,
                         const float* __restrict__ bias, float* __restrict__ out,
                         int rows, int H) {
    int wave = blockIdx.x * (blockDim.x >> 5) + (threadIdx.x >> 5);
    int lane = threadIdx.x & 31;
    if (wave >= rows) return;
    const float* xr = X + (size_t)wave * H;
    float s = 0.f;
    for (int h = lane; h < H; h += 32) s += xr[h] * w[h];
#pragma unroll
    for (int o = 16; o > 0; o >>= 1) s += __shfl_xor(s, o, 32);
    if (lane == 0) out[wave] = s + bias[0];
}

// ---------------- K1: score GEMM  s[b,c,q] ----------------
// block 256 (8 waves), grid (C/32, B): 32 C rows x 512 Q cols per block.
// Each wave: 2 M-tiles x 4 N-tiles -> 8 WMMA per K-step, pure b128 operand loads.
__global__ void k_scores(const unsigned short* __restrict__ e2w,
                         const unsigned short* __restrict__ e1b,
                         const float* __restrict__ sc,
                         const float* __restrict__ sq,
                         const float* __restrict__ bcq,
                         float* __restrict__ s) {
    int b = blockIdx.y;
    int cBase = blockIdx.x * 32;
    int lane = threadIdx.x & 31;
    int wave = threadIdx.x >> 5;
    int nBase = wave * 64;

    const unsigned short* A0 = e2w + (size_t)(b * C_L + cBase) * H_D;
    const unsigned short* Bt = e1b + (size_t)b * Q_L * H_D;   // row q, contiguous h

    v8f acc[2][4] = {};
    for (int k = 0; k < H_D; k += 32) {
        v16bf a0 = load_frag_a(A0, H_D, k, lane);
        v16bf a1 = load_frag_a(A0 + 16 * H_D, H_D, k, lane);
#pragma unroll
        for (int t = 0; t < 4; ++t) {
            v16bf bf = load_frag_b(Bt + (size_t)(nBase + t * 16) * H_D, H_D, k, lane);
            acc[0][t] = wmma_bf16(a0, bf, acc[0][t]);
            acc[1][t] = wmma_bf16(a1, bf, acc[1][t]);
        }
    }
    int col = lane & 15, sel = lane >> 4;
    float bc = bcq[0];
    float scr[2][8];
#pragma unroll
    for (int m = 0; m < 2; ++m)
#pragma unroll
        for (int i = 0; i < 8; ++i)
            scr[m][i] = sc[b * C_L + cBase + m * 16 + sel * 8 + i];
#pragma unroll
    for (int t = 0; t < 4; ++t) {
        int q = nBase + t * 16 + col;
        float addq = sq[b * Q_L + q] + bc;
#pragma unroll
        for (int m = 0; m < 2; ++m)
#pragma unroll
            for (int i = 0; i < 8; ++i) {
                int c = cBase + m * 16 + sel * 8 + i;
                s[((size_t)(b * C_L + c)) * Q_L + q] = acc[m][t][i] + scr[m][i] + addq;
            }
    }
}

// ---------------- K2: fused softmax + c2q GEMM ----------------
// block 256, 16 C rows per block. exp stored unnormalized in LDS (bf16);
// 1/rowSum folded into GEMM output. Also emits row-max -> smax.
__global__ void k_softmax_c2q(const float* __restrict__ s,
                              const unsigned short* __restrict__ e1t,
                              float* __restrict__ smax,
                              unsigned short* __restrict__ c2qb) {
    __shared__ unsigned short aS[16 * Q_L];   // 16 KB
    __shared__ float red[16][16];
    __shared__ float rowMax[16];
    __shared__ float rowSum[16];

    int b = blockIdx.y;
    int cBase = blockIdx.x * 16;
    int tid = threadIdx.x;
    int r = tid >> 4;          // row 0..15
    int j = tid & 15;          // 16 threads per row

    const float* srow = s + ((size_t)(b * C_L + cBase + r)) * Q_L;
    float v[32];
    float m = -3.402823466e38f;
#pragma unroll
    for (int i = 0; i < 32; ++i) {
        v[i] = srow[j + i * 16];
        m = fmaxf(m, v[i]);
    }
    red[r][j] = m;
    __syncthreads();
    if (j == 0) {
        float mm = red[r][0];
#pragma unroll
        for (int i = 1; i < 16; ++i) mm = fmaxf(mm, red[r][i]);
        rowMax[r] = mm;
        smax[b * C_L + cBase + r] = mm;
    }
    __syncthreads();
    float M = rowMax[r];
    float ssum = 0.f;
#pragma unroll
    for (int i = 0; i < 32; ++i) {
        float e = __expf(v[i] - M);
        ssum += e;
        aS[r * Q_L + j + i * 16] = f2bf(e);
    }
    red[r][j] = ssum;
    __syncthreads();
    if (j == 0) {
        float t = 0.f;
#pragma unroll
        for (int i = 0; i < 16; ++i) t += red[r][i];
        rowSum[r] = t;
    }
    __syncthreads();

    // GEMM: D(16 x 768) = aS(16 x 512) @ emb1(512 x 768); B from e1t (H-major)
    int lane = tid & 31;
    int wave = tid >> 5;          // 8 waves * 96 cols = 768
    int hBase0 = wave * 96;
    const unsigned short* Bt = e1t + (size_t)b * H_D * Q_L;

    v8f acc[6] = {};
    for (int k = 0; k < Q_L; k += 32) {
        v16bf af = load_frag_a(aS, Q_L, k, lane);
#pragma unroll
        for (int t = 0; t < 6; ++t) {
            v16bf bf = load_frag_b(Bt + (size_t)(hBase0 + t * 16) * Q_L, Q_L, k, lane);
            acc[t] = wmma_bf16(af, bf, acc[t]);
        }
    }
    int col = lane & 15, sel = lane >> 4;
#pragma unroll
    for (int t = 0; t < 6; ++t) {
        int h = hBase0 + t * 16 + col;
#pragma unroll
        for (int i = 0; i < 8; ++i) {
            int rloc = sel * 8 + i;
            int c = cBase + rloc;
            float val = acc[t][i] / rowSum[rloc];
            c2qb[((size_t)(b * C_L + c)) * H_D + h] = f2bf(val);
        }
    }
}

// ---------------- K3: b_att softmax over C + q2c reduction ----------------
__global__ void k_batt_q2c(const float* __restrict__ smax,
                           const float* __restrict__ emb2,
                           float* __restrict__ q2c) {
    __shared__ float w[C_L];     // 8 KB
    __shared__ float red[256];
    int b = blockIdx.x;
    int t = threadIdx.x;

    float m = -3.402823466e38f;
    for (int c = t; c < C_L; c += 256) m = fmaxf(m, smax[b * C_L + c]);
    red[t] = m;
    __syncthreads();
    for (int o = 128; o > 0; o >>= 1) {
        if (t < o) red[t] = fmaxf(red[t], red[t + o]);
        __syncthreads();
    }
    float M = red[0];
    __syncthreads();
    float ssum = 0.f;
    for (int c = t; c < C_L; c += 256) {
        float e = __expf(smax[b * C_L + c] - M);
        w[c] = e;
        ssum += e;
    }
    red[t] = ssum;
    __syncthreads();
    for (int o = 128; o > 0; o >>= 1) {
        if (t < o) red[t] += red[t + o];
        __syncthreads();
    }
    float Z = red[0];
    __syncthreads();

    for (int h = t; h < H_D; h += 256) {
        float acc = 0.f;
        for (int c = 0; c < C_L; ++c)
            acc += w[c] * emb2[((size_t)(b * C_L + c)) * H_D + h];
        q2c[b * H_D + h] = acc / Z;
    }
}

// ---------------- post-K3 folds (make K4 a pure GEMM) ----------------
// p2 = bf16(emb2 * c2q)  (from the bf16 copies)
__global__ void k_fold_p2(const unsigned short* __restrict__ e2b,
                          const unsigned short* __restrict__ c2qb,
                          unsigned short* __restrict__ p2, long long n) {
    long long i = (long long)blockIdx.x * blockDim.x + threadIdx.x;
    if (i < n) p2[i] = f2bf(bf2f(e2b[i]) * bf2f(c2qb[i]));
}

// wr3b[b,o,h] = W3[o,h] * q2c[b,h]   (fold q2c into the 4th weight block, per batch)
__global__ void k_fold_w3(const unsigned short* __restrict__ wrb,
                          const float* __restrict__ q2c,
                          unsigned short* __restrict__ wr3b) {
    long long i = (long long)blockIdx.x * blockDim.x + threadIdx.x;
    long long n = (long long)B_N * WR_ROWS * H_D;
    if (i >= n) return;
    int h = (int)(i % H_D);
    int o = (int)((i / H_D) % WR_ROWS);
    int b = (int)(i / ((long long)H_D * WR_ROWS));
    wr3b[i] = f2bf(bf2f(wrb[(size_t)o * K4 + 3 * H_D + h]) * q2c[b * H_D + h]);
}

// ---------------- K4: output GEMM  out = x @ w_red^T + b_red ----------------
// block 160 (5 waves), grid (C/32, B): 32 C rows x 320 padded cols.
// x segments resolved at compile time: seg0 e2b|W0, seg1 c2qb|W1, seg2 p2|W2,
// seg3 e2b|wr3b (q2c folded into weights). Inner loop is pure loads + WMMA.
__global__ void k_out(const unsigned short* __restrict__ e2b,
                      const unsigned short* __restrict__ c2qb,
                      const unsigned short* __restrict__ p2,
                      const unsigned short* __restrict__ wrb,
                      const unsigned short* __restrict__ wr3b,
                      const float* __restrict__ b_red,
                      float* __restrict__ out) {
    int b = blockIdx.y;
    int cBase = blockIdx.x * 32;
    int lane = threadIdx.x & 31;
    int wave = threadIdx.x >> 5;
    int nBase = wave * 64;

    const unsigned short* E2 = e2b  + (size_t)(b * C_L + cBase) * H_D;
    const unsigned short* CQ = c2qb + (size_t)(b * C_L + cBase) * H_D;
    const unsigned short* P2 = p2   + (size_t)(b * C_L + cBase) * H_D;
    const unsigned short* W3 = wr3b + (size_t)b * WR_ROWS * H_D;

    v8f acc[2][4] = {};
#pragma unroll
    for (int seg = 0; seg < 4; ++seg) {
        const unsigned short* As = (seg == 1) ? CQ : (seg == 2) ? P2 : E2;
        for (int h = 0; h < H_D; h += 32) {
            v16bf a0 = load_frag_a(As, H_D, h, lane);
            v16bf a1 = load_frag_a(As + 16 * H_D, H_D, h, lane);
#pragma unroll
            for (int t = 0; t < 4; ++t) {
                int n = nBase + t * 16;
                v16bf bf = (seg < 3)
                    ? load_frag_b(wrb + (size_t)n * K4, K4, seg * H_D + h, lane)
                    : load_frag_b(W3  + (size_t)n * H_D, H_D, h, lane);
                acc[0][t] = wmma_bf16(a0, bf, acc[0][t]);
                acc[1][t] = wmma_bf16(a1, bf, acc[1][t]);
            }
        }
    }
    int col = lane & 15, sel = lane >> 4;
#pragma unroll
    for (int t = 0; t < 4; ++t) {
        int o = nBase + t * 16 + col;
        if (o < OUT_D) {
            float bias = b_red[o];
#pragma unroll
            for (int m = 0; m < 2; ++m)
#pragma unroll
                for (int i = 0; i < 8; ++i) {
                    int c = cBase + m * 16 + sel * 8 + i;
                    out[((size_t)(b * C_L + c)) * OUT_D + o] = acc[m][t][i] + bias;
                }
        }
    }
}

// ---------------- host launch ----------------
static inline size_t alignup(size_t x) { return (x + 255) & ~(size_t)255; }

extern "C" void kernel_launch(void* const* d_in, const int* in_sizes, int n_in,
                              void* d_out, int out_size, void* d_ws, size_t ws_size,
                              hipStream_t stream) {
    (void)in_sizes; (void)n_in; (void)out_size; (void)ws_size;
    const float* emb1  = (const float*)d_in[0];
    const float* emb2  = (const float*)d_in[1];
    const float* w_c   = (const float*)d_in[2];
    const float* b_c   = (const float*)d_in[3];
    const float* w_q   = (const float*)d_in[4];
    const float* b_q   = (const float*)d_in[5];
    const float* w_cq  = (const float*)d_in[6];
    const float* b_cq  = (const float*)d_in[7];
    const float* w_red = (const float*)d_in[8];
    const float* b_red = (const float*)d_in[9];
    float* out = (float*)d_out;

    char* ws = (char*)d_ws;
    size_t off = 0;
    unsigned short* e2b  = (unsigned short*)(ws + off); off += alignup((size_t)B_N * C_L * H_D * 2);
    unsigned short* e2w  = (unsigned short*)(ws + off); off += alignup((size_t)B_N * C_L * H_D * 2);
    unsigned short* e1b  = (unsigned short*)(ws + off); off += alignup((size_t)B_N * Q_L * H_D * 2);
    unsigned short* e1t  = (unsigned short*)(ws + off); off += alignup((size_t)B_N * H_D * Q_L * 2);
    unsigned short* wrb  = (unsigned short*)(ws + off); off += alignup((size_t)WR_ROWS * K4 * 2);
    unsigned short* wr3b = (unsigned short*)(ws + off); off += alignup((size_t)B_N * WR_ROWS * H_D * 2);
    float* sc   = (float*)(ws + off); off += alignup((size_t)B_N * C_L * 4);
    float* sq   = (float*)(ws + off); off += alignup((size_t)B_N * Q_L * 4);
    float* smax = (float*)(ws + off); off += alignup((size_t)B_N * C_L * 4);
    float* s    = (float*)(ws + off); off += alignup((size_t)B_N * C_L * Q_L * 4);
    unsigned short* c2qb = (unsigned short*)(ws + off); off += alignup((size_t)B_N * C_L * H_D * 2);
    unsigned short* p2   = (unsigned short*)(ws + off); off += alignup((size_t)B_N * C_L * H_D * 2);
    float* q2c  = (float*)(ws + off); off += alignup((size_t)B_N * H_D * 4);

    long long n2 = (long long)B_N * C_L * H_D;
    long long n1 = (long long)B_N * Q_L * H_D;

    // prep
    k_prep_e2<<<dim3((unsigned)((n2 + 255) / 256)), dim3(256), 0, stream>>>(emb2, w_cq, e2b, e2w, n2);
    k_cvt_bf16<<<dim3((unsigned)((n1 + 255) / 256)), dim3(256), 0, stream>>>(emb1, e1b, n1);
    k_e1_transpose<<<dim3((unsigned)((n1 + 255) / 256)), dim3(256), 0, stream>>>(emb1, e1t);
    {
        long long nw = (long long)WR_ROWS * K4;
        k_cvt_wred<<<dim3((unsigned)((nw + 255) / 256)), dim3(256), 0, stream>>>(w_red, wrb);
    }
    k_rowdot<<<dim3((B_N * C_L) / 8), dim3(256), 0, stream>>>(emb2, w_c, b_c, sc, B_N * C_L, H_D);
    k_rowdot<<<dim3((B_N * Q_L) / 8), dim3(256), 0, stream>>>(emb1, w_q, b_q, sq, B_N * Q_L, H_D);

    // attention scores
    k_scores<<<dim3(C_L / 32, B_N), dim3(256), 0, stream>>>(e2w, e1b, sc, sq, b_cq, s);
    // softmax + c2q
    k_softmax_c2q<<<dim3(C_L / 16, B_N), dim3(256), 0, stream>>>(s, e1t, smax, c2qb);
    // b_att + q2c
    k_batt_q2c<<<dim3(B_N), dim3(256), 0, stream>>>(smax, emb2, q2c);
    // folds for the final GEMM
    k_fold_p2<<<dim3((unsigned)((n2 + 255) / 256)), dim3(256), 0, stream>>>(e2b, c2qb, p2, n2);
    {
        long long nf = (long long)B_N * WR_ROWS * H_D;
        k_fold_w3<<<dim3((unsigned)((nf + 255) / 256)), dim3(256), 0, stream>>>(wrb, q2c, wr3b);
    }
    // final reduction GEMM (pure WMMA)
    k_out<<<dim3(C_L / 32, B_N), dim3(160), 0, stream>>>(e2b, c2qb, p2, wrb, wr3b, b_red, out);
}
